// TransforMerBlock_1460288880896
// MI455X (gfx1250) — matrix operand order repaired and checked by reading, not compiled, not run on hardware
//
#include <hip/hip_runtime.h>
#include <stdint.h>

#define BB   4
#define SS   2048
#define DD   1024
#define HH   16
#define HDIM 64
#define DFFN 4096
#define BSTOK (BB*SS)

typedef __attribute__((ext_vector_type(16))) __bf16 v16bf;
typedef __attribute__((ext_vector_type(8)))  __bf16 v8bf;
typedef __attribute__((ext_vector_type(8)))  float  v8f;

__device__ __forceinline__ unsigned short f2bf(float f) {
  union { float f; unsigned u; } v; v.f = f;
  unsigned r = (v.u + 0x7FFFu + ((v.u >> 16) & 1u)) >> 16;
  return (unsigned short)r;
}
__device__ __forceinline__ float bf2f(unsigned short h) {
  union { unsigned u; float f; } v; v.u = ((unsigned)h) << 16;
  return v.f;
}
__device__ __forceinline__ v16bf cat8(v8bf lo, v8bf hi) {
  return __builtin_shufflevector(lo, hi, 0,1,2,3,4,5,6,7,8,9,10,11,12,13,14,15);
}

// Async DMA: global (128b per lane) -> LDS, tracked by ASYNCcnt (no VGPR data).
// vdst = 32-bit LDS byte address, vaddr = 64-bit global address.
__device__ __forceinline__ void async_ld_b128(void* lds, const void* g) {
  const unsigned l = (unsigned)(uintptr_t)lds;          // low 32 bits == AS3 offset
  const unsigned long long a = (unsigned long long)(uintptr_t)g;
  asm volatile("global_load_async_to_lds_b128 %0, %1, off"
               :: "v"(l), "v"(a) : "memory");
}

// ---------------------------------------------------------------------------
// f32 -> bf16 convert (weights)
// ---------------------------------------------------------------------------
__global__ void k_cvt_bf16(const float* __restrict__ in,
                           unsigned short* __restrict__ out, int n) {
  for (int i = blockIdx.x * blockDim.x + threadIdx.x; i < n;
       i += gridDim.x * blockDim.x)
    out[i] = f2bf(in[i]);
}

// ---------------------------------------------------------------------------
// RMSNorm: f32 [rows, DD] -> bf16 [rows, DD]; one block (8 waves) per row
// ---------------------------------------------------------------------------
__global__ __launch_bounds__(256) void k_rmsnorm(const float* __restrict__ x,
                                                 const float* __restrict__ w,
                                                 unsigned short* __restrict__ out) {
  const int row = blockIdx.x;
  const int tid = threadIdx.x;
  const float* xr = x + (size_t)row * DD;
  float lx[4];
  float ss = 0.f;
#pragma unroll
  for (int i = 0; i < 4; ++i) { lx[i] = xr[tid + i * 256]; ss += lx[i] * lx[i]; }
#pragma unroll
  for (int m = 16; m >= 1; m >>= 1) ss += __shfl_xor(ss, m, 32);
  __shared__ float red[8];
  if ((tid & 31) == 0) red[tid >> 5] = ss;
  __syncthreads();
  float tot = 0.f;
#pragma unroll
  for (int i = 0; i < 8; ++i) tot += red[i];
  const float inv = rsqrtf(tot * (1.0f / DD) + 1e-5f);
#pragma unroll
  for (int i = 0; i < 4; ++i) {
    const int e = tid + i * 256;
    out[(size_t)row * DD + e] = f2bf(lx[i] * inv * w[e]);
  }
}

// ---------------------------------------------------------------------------
// WMMA GEMM: C[M,N] = A[M,K](bf16) * B[N,K](bf16)^T
// mode 0: Cf = acc      mode 1: Cf = acc + R      mode 2: Cb = bf16(acc)
// Block tile 128x128, K-step 32, 8 waves (2x4), wave tile 64x32 (4x2 WMMAs).
// Staging: double-buffered LDS filled with global_load_async_to_lds_b128,
// pipelined one K-tile ahead, synchronized with s_wait_asynccnt.
// ---------------------------------------------------------------------------
#define LDT 48  // LDS row stride in bf16 elems (96B: 32B-aligned, conflict-skewed)

__global__ __launch_bounds__(256) void k_gemm_bf16(
    const unsigned short* __restrict__ A, const unsigned short* __restrict__ Bw,
    int M, int N, int K, float* __restrict__ Cf, unsigned short* __restrict__ Cb,
    const float* __restrict__ R, int mode) {
  __shared__ __attribute__((aligned(64))) unsigned short lA[2][128 * LDT];
  __shared__ __attribute__((aligned(64))) unsigned short lB[2][128 * LDT];

  const int tid  = threadIdx.x;
  const int lane = tid & 31;
  const int wv   = tid >> 5;
  const int half = lane >> 4;
  const int lm   = lane & 15;
  const int wrow = wv >> 2;  // 0..1
  const int wcol = wv & 3;   // 0..3
  const int bm = blockIdx.y * 128;
  const int bn = blockIdx.x * 128;

  const v8f vzero = {0.f, 0.f, 0.f, 0.f, 0.f, 0.f, 0.f, 0.f};
  v8f acc[4][2];
#pragma unroll
  for (int i = 0; i < 4; ++i)
#pragma unroll
    for (int j = 0; j < 2; ++j) acc[i][j] = vzero;

  const int r0 = tid >> 2;           // 0..63
  const int c0 = (tid & 3) * 8;      // 0,8,16,24

  // issue one K-tile's async copies (4 b128 per thread) into buffer `buf`
  auto issue_tile = [&](int kt, int buf) {
    async_ld_b128(&lA[buf][(r0)      * LDT + c0], A  + (size_t)(bm + r0)      * K + kt + c0);
    async_ld_b128(&lA[buf][(r0 + 64) * LDT + c0], A  + (size_t)(bm + r0 + 64) * K + kt + c0);
    async_ld_b128(&lB[buf][(r0)      * LDT + c0], Bw + (size_t)(bn + r0)      * K + kt + c0);
    async_ld_b128(&lB[buf][(r0 + 64) * LDT + c0], Bw + (size_t)(bn + r0 + 64) * K + kt + c0);
  };

  const int ntk = K >> 5;
  issue_tile(0, 0);

  for (int t = 0; t < ntk; ++t) {
    const int cur = t & 1;
    if (t + 1 < ntk) {
      issue_tile((t + 1) << 5, (t + 1) & 1);
      if (t + 2 < ntk) {
        __builtin_prefetch(A  + (size_t)(bm + r0) * K + ((t + 2) << 5) + c0, 0, 1);
        __builtin_prefetch(Bw + (size_t)(bn + r0) * K + ((t + 2) << 5) + c0, 0, 1);
      }
      // in-order async completion: <=4 outstanding => tile t fully in LDS
      asm volatile("s_wait_asynccnt 4" ::: "memory");
    } else {
      asm volatile("s_wait_asynccnt 0" ::: "memory");
    }
    __syncthreads();

    v16bf af[4], bfg[2];
#pragma unroll
    for (int i = 0; i < 4; ++i) {
      const __bf16* p = (const __bf16*)&lA[cur][(wrow * 64 + i * 16 + lm) * LDT];
      af[i] = cat8(*(const v8bf*)(p + half * 8), *(const v8bf*)(p + 16 + half * 8));
    }
#pragma unroll
    for (int j = 0; j < 2; ++j) {
      const __bf16* p = (const __bf16*)&lB[cur][(wcol * 32 + j * 16 + lm) * LDT];
      bfg[j] = cat8(*(const v8bf*)(p + half * 8), *(const v8bf*)(p + 16 + half * 8));
    }
#pragma unroll
    for (int i = 0; i < 4; ++i)
#pragma unroll
      for (int j = 0; j < 2; ++j)
        acc[i][j] = __builtin_amdgcn_wmma_f32_16x16x32_bf16(
            false, af[i], false, bfg[j], (short)0, acc[i][j], false, false);

    __syncthreads();  // all waves done reading `cur` before it is refilled
  }

#pragma unroll
  for (int i = 0; i < 4; ++i)
#pragma unroll
    for (int j = 0; j < 2; ++j) {
      const int n = bn + wcol * 32 + j * 16 + lm;
#pragma unroll
      for (int r = 0; r < 8; ++r) {
        const int m = bm + wrow * 64 + i * 16 + 8 * half + r;
        const size_t idx = (size_t)m * N + n;
        const float v = acc[i][j][r];
        if (mode == 0)      Cf[idx] = v;
        else if (mode == 1) Cf[idx] = v + R[idx];
        else                Cb[idx] = f2bf(v);
      }
    }
}

// ---------------------------------------------------------------------------
// RoPE + per-head repack: q,k,v bf16 [tok, DD] -> qh,kh bf16 [b,h,s,d],
// vT bf16 [b,h,d,s]. One thread per (even,odd) pair.
// ---------------------------------------------------------------------------
__global__ __launch_bounds__(256) void k_rope_pack(
    const unsigned short* __restrict__ q, const unsigned short* __restrict__ k,
    const unsigned short* __restrict__ v, unsigned short* __restrict__ qh,
    unsigned short* __restrict__ kh, unsigned short* __restrict__ vt) {
  const int p = blockIdx.x * blockDim.x + threadIdx.x;  // B*S*D/2
  const int d2 = p & 31;
  const int h  = (p >> 5) & (HH - 1);
  const int s  = (p >> 9) & (SS - 1);
  const int b  = p >> 20;
  const size_t src = (size_t)(b * SS + s) * DD + h * HDIM + 2 * d2;
  const float fr = __powf(10000.0f, -(float)d2 * (1.0f / 32.0f));
  float sn, cs;
  __sincosf((float)s * fr, &sn, &cs);
  const int bh = b * HH + h;
  {
    const float e = bf2f(q[src]), o = bf2f(q[src + 1]);
    const size_t dst = ((size_t)bh * SS + s) * HDIM + 2 * d2;
    qh[dst]     = f2bf(cs * e - sn * o);
    qh[dst + 1] = f2bf(cs * o + sn * e);
  }
  {
    const float e = bf2f(k[src]), o = bf2f(k[src + 1]);
    const size_t dst = ((size_t)bh * SS + s) * HDIM + 2 * d2;
    kh[dst]     = f2bf(cs * e - sn * o);
    kh[dst + 1] = f2bf(cs * o + sn * e);
  }
  {
    const float e = bf2f(v[src]), o = bf2f(v[src + 1]);
    const size_t dst = ((size_t)bh * HDIM + 2 * d2) * SS + s;
    vt[dst]      = f2bf(e);
    vt[dst + SS] = f2bf(o);
  }
}

// ---------------------------------------------------------------------------
// Causal flash attention. One workgroup = one (b,h, 128-row Q block):
// 8 waves x 16 Q rows. K (32x64) and V^T (64x32) tiles are staged in LDS by
// the async DMA engine (double buffered, ASYNCcnt-pipelined); all 8 waves
// build WMMA fragments from LDS. Online softmax via 16-lane shuffles; P is
// restaged through per-wave LDS as an A-fragment for the PV WMMAs.
// LDS strides: K rows 80 elems (160B), V rows 48 elems (96B) -> 32B-aligned,
// bank-skewed.
// ---------------------------------------------------------------------------
__global__ __launch_bounds__(256) void k_flash_attn(
    const unsigned short* __restrict__ qh, const unsigned short* __restrict__ kh,
    const unsigned short* __restrict__ vt, unsigned short* __restrict__ attn) {
  __shared__ __attribute__((aligned(64))) unsigned short ldsK[2][32 * 80];
  __shared__ __attribute__((aligned(64))) unsigned short ldsV[2][64 * 48];
  __shared__ __attribute__((aligned(64))) unsigned short pls[8][16 * 40];

  const int tid  = threadIdx.x;
  const int lane = tid & 31;
  const int wv   = tid >> 5;
  const int half = lane >> 4;
  const int lm   = lane & 15;
  const int bh   = blockIdx.x >> 4;        // B*H = 64 heads
  const int qblk = blockIdx.x & 15;        // S/128 = 16 Q blocks
  const int q0   = qblk * 128 + wv * 16;   // this wave's 16 Q rows

  const __bf16* qp = (const __bf16*)qh + ((size_t)bh * SS + q0 + lm) * HDIM;
  const v16bf qa0 = *(const v16bf*)(qp + half * 16);
  const v16bf qa1 = *(const v16bf*)(qp + 32 + half * 16);

  const v8f vzero = {0.f, 0.f, 0.f, 0.f, 0.f, 0.f, 0.f, 0.f};
  float mst[8], lst[8];
  v8f o[4];
#pragma unroll
  for (int r = 0; r < 8; ++r) { mst[r] = -3.0e38f; lst[r] = 0.f; }
#pragma unroll
  for (int dt = 0; dt < 4; ++dt) o[dt] = vzero;

  unsigned short* pw = pls[wv];

  // per-thread async staging slots: K tile 32x64 (256 x b128), V^T tile 64x32
  const int kr = tid >> 3, kc = (tid & 7) * 8;   // K: row 0..31, col chunk
  const int vr = tid >> 2, vc = (tid & 3) * 8;   // V^T: row 0..63, col chunk
  auto issue_kv = [&](int kb, int buf) {
    async_ld_b128(&ldsK[buf][kr * 80 + kc],
                  kh + ((size_t)bh * SS + kb + kr) * HDIM + kc);
    async_ld_b128(&ldsV[buf][vr * 48 + vc],
                  vt + ((size_t)bh * HDIM + vr) * SS + kb + vc);
  };

  const int nt = qblk * 4 + 4;   // 32-col K tiles covering rows 0..q0_max+15
  issue_kv(0, 0);

  for (int t = 0; t < nt; ++t) {
    const int cur = t & 1;
    const int kb0 = t * 32;
    if (t + 1 < nt) {
      issue_kv((t + 1) * 32, (t + 1) & 1);
      asm volatile("s_wait_asynccnt 2" ::: "memory");   // in-order: tile t done
    } else {
      asm volatile("s_wait_asynccnt 0" ::: "memory");
    }
    __syncthreads();

    if (kb0 <= q0 + 15) {   // causal: this wave still has unmasked columns
      const __bf16* kb = (const __bf16*)ldsK[cur];
      const __bf16* kp0 = kb + lm * 80 + half * 16;
      const __bf16* kp1 = kb + (16 + lm) * 80 + half * 16;

      v8f s0 = __builtin_amdgcn_wmma_f32_16x16x32_bf16(
          false, qa0, false, *(const v16bf*)kp0, (short)0, vzero, false, false);
      s0 = __builtin_amdgcn_wmma_f32_16x16x32_bf16(
          false, qa1, false, *(const v16bf*)(kp0 + 32), (short)0, s0, false, false);
      v8f s1 = __builtin_amdgcn_wmma_f32_16x16x32_bf16(
          false, qa0, false, *(const v16bf*)kp1, (short)0, vzero, false, false);
      s1 = __builtin_amdgcn_wmma_f32_16x16x32_bf16(
          false, qa1, false, *(const v16bf*)(kp1 + 32), (short)0, s1, false, false);

      float sa[8], sb[8];
#pragma unroll
      for (int r = 0; r < 8; ++r) { sa[r] = s0[r] * 0.125f; sb[r] = s1[r] * 0.125f; }
      if (kb0 + 31 > q0) {   // tile touches the diagonal for this wave
        const int ca = kb0 + lm, cb = ca + 16;
#pragma unroll
        for (int r = 0; r < 8; ++r) {
          const int row = q0 + r + 8 * half;
          if (ca > row) sa[r] = -1.0e9f;
          if (cb > row) sb[r] = -1.0e9f;
        }
      }

#pragma unroll
      for (int r = 0; r < 8; ++r) {
        float tm = fmaxf(sa[r], sb[r]);
        tm = fmaxf(tm, __shfl_xor(tm, 1, 32));
        tm = fmaxf(tm, __shfl_xor(tm, 2, 32));
        tm = fmaxf(tm, __shfl_xor(tm, 4, 32));
        tm = fmaxf(tm, __shfl_xor(tm, 8, 32));
        const float mn = fmaxf(mst[r], tm);
        const float al = __expf(mst[r] - mn);
        mst[r] = mn;
        const float p0 = __expf(sa[r] - mn);
        const float p1 = __expf(sb[r] - mn);
        float rs = p0 + p1;
        rs += __shfl_xor(rs, 1, 32);
        rs += __shfl_xor(rs, 2, 32);
        rs += __shfl_xor(rs, 4, 32);
        rs += __shfl_xor(rs, 8, 32);
        lst[r] = lst[r] * al + rs;
#pragma unroll
        for (int dt = 0; dt < 4; ++dt) o[dt][r] *= al;
        pw[(r + 8 * half) * 40 + lm]      = f2bf(p0);
        pw[(r + 8 * half) * 40 + 16 + lm] = f2bf(p1);
      }

      // wave-local LDS visibility: DS ops are in-order per wave; fence the
      // compiler and drain DScnt before the cross-lane fragment reads.
      __builtin_amdgcn_wave_barrier();
      asm volatile("s_wait_dscnt 0" ::: "memory");
      __builtin_amdgcn_wave_barrier();

      const __bf16* pr = (const __bf16*)pw + lm * 40;
      const v16bf pa = cat8(*(const v8bf*)(pr + half * 8),
                            *(const v8bf*)(pr + 16 + half * 8));
#pragma unroll
      for (int dt = 0; dt < 4; ++dt) {
        const __bf16* vp = (const __bf16*)ldsV[cur] + (dt * 16 + lm) * 48;
        o[dt] = __builtin_amdgcn_wmma_f32_16x16x32_bf16(
            false, pa, false, cat8(*(const v8bf*)(vp + half * 8), *(const v8bf*)(vp + 16 + half * 8)), (short)0, o[dt], false, false);
      }
    }

    __syncthreads();   // all waves done reading `cur` before refill
  }

  const int b = bh >> 4;
  const int h = bh & (HH - 1);
#pragma unroll
  for (int r = 0; r < 8; ++r) {
    const float inv = 1.0f / lst[r];
    const int row = q0 + r + 8 * half;
#pragma unroll
    for (int dt = 0; dt < 4; ++dt) {
      attn[(size_t)(b * SS + row) * DD + h * HDIM + dt * 16 + lm] =
          f2bf(o[dt][r] * inv);
    }
  }
}

// ---------------------------------------------------------------------------
// SwiGLU elementwise: out = silu(g) * up, bf16 in/out
// ---------------------------------------------------------------------------
__global__ void k_silu_mul(const unsigned short* __restrict__ g,
                           const unsigned short* __restrict__ u,
                           unsigned short* __restrict__ out, int n) {
  for (int i = blockIdx.x * blockDim.x + threadIdx.x; i < n;
       i += gridDim.x * blockDim.x) {
    const float gv = bf2f(g[i]);
    const float uv = bf2f(u[i]);
    out[i] = f2bf(gv / (1.f + __expf(-gv)) * uv);
  }
}

// ---------------------------------------------------------------------------
extern "C" void kernel_launch(void* const* d_in, const int* in_sizes, int n_in,
                              void* d_out, int out_size, void* d_ws, size_t ws_size,
                              hipStream_t stream) {
  (void)in_sizes; (void)n_in; (void)out_size; (void)ws_size;
  const float* x   = (const float*)d_in[0];
  const float* wq  = (const float*)d_in[1];
  const float* wk  = (const float*)d_in[2];
  const float* wv  = (const float*)d_in[3];
  const float* wo  = (const float*)d_in[4];
  const float* ln1 = (const float*)d_in[5];
  const float* ln2 = (const float*)d_in[6];
  const float* w1  = (const float*)d_in[7];
  const float* w2  = (const float*)d_in[8];
  const float* w3  = (const float*)d_in[9];
  float* out = (float*)d_out;

  char* base = (char*)d_ws;
  size_t off = 0;
  auto alloc = [&](size_t bytes) -> void* {
    void* p = base + off;
    off += (bytes + 255) & ~(size_t)255;
    return p;
  };
  const size_t nDxD  = (size_t)DD * DD;
  const size_t nFxD  = (size_t)DFFN * DD;
  const size_t nTok  = (size_t)BSTOK * DD;
  const size_t nTokF = (size_t)BSTOK * DFFN;

  unsigned short* wqb = (unsigned short*)alloc(nDxD * 2);
  unsigned short* wkb = (unsigned short*)alloc(nDxD * 2);
  unsigned short* wvb = (unsigned short*)alloc(nDxD * 2);
  unsigned short* wob = (unsigned short*)alloc(nDxD * 2);
  unsigned short* w1b = (unsigned short*)alloc(nFxD * 2);
  unsigned short* w2b = (unsigned short*)alloc(nFxD * 2);
  unsigned short* w3b = (unsigned short*)alloc(nFxD * 2);
  unsigned short* h1  = (unsigned short*)alloc(nTok * 2);
  unsigned short* qtk = (unsigned short*)alloc(nTok * 2);
  unsigned short* ktk = (unsigned short*)alloc(nTok * 2);
  unsigned short* vtk = (unsigned short*)alloc(nTok * 2);
  unsigned short* qhd = (unsigned short*)alloc(nTok * 2);
  unsigned short* khd = (unsigned short*)alloc(nTok * 2);
  unsigned short* vtr = (unsigned short*)alloc(nTok * 2);
  unsigned short* atn = (unsigned short*)alloc(nTok * 2);
  float*          x2  = (float*)alloc(nTok * 4);
  unsigned short* h2  = (unsigned short*)alloc(nTok * 2);
  unsigned short* gg  = (unsigned short*)alloc(nTokF * 2);
  unsigned short* uu  = (unsigned short*)alloc(nTokF * 2);
  unsigned short* fi  = (unsigned short*)alloc(nTokF * 2);

  const dim3 blk(256);

  // weights -> bf16
  k_cvt_bf16<<<2048, blk, 0, stream>>>(wq, wqb, (int)nDxD);
  k_cvt_bf16<<<2048, blk, 0, stream>>>(wk, wkb, (int)nDxD);
  k_cvt_bf16<<<2048, blk, 0, stream>>>(wv, wvb, (int)nDxD);
  k_cvt_bf16<<<2048, blk, 0, stream>>>(wo, wob, (int)nDxD);
  k_cvt_bf16<<<4096, blk, 0, stream>>>(w1, w1b, (int)nFxD);
  k_cvt_bf16<<<4096, blk, 0, stream>>>(w2, w2b, (int)nFxD);
  k_cvt_bf16<<<4096, blk, 0, stream>>>(w3, w3b, (int)nFxD);

  // h1 = rmsnorm(x) * ln1
  k_rmsnorm<<<BSTOK, blk, 0, stream>>>(x, ln1, h1);

  // q,k,v projections (bf16 out)
  k_gemm_bf16<<<dim3(DD / 128, BSTOK / 128), blk, 0, stream>>>(
      h1, wqb, BSTOK, DD, DD, nullptr, qtk, nullptr, 2);
  k_gemm_bf16<<<dim3(DD / 128, BSTOK / 128), blk, 0, stream>>>(
      h1, wkb, BSTOK, DD, DD, nullptr, ktk, nullptr, 2);
  k_gemm_bf16<<<dim3(DD / 128, BSTOK / 128), blk, 0, stream>>>(
      h1, wvb, BSTOK, DD, DD, nullptr, vtk, nullptr, 2);

  // rope + per-head repack (+ V transpose)
  k_rope_pack<<<(BB * SS * DD / 2) / 256, blk, 0, stream>>>(qtk, ktk, vtk, qhd,
                                                           khd, vtr);

  // causal flash attention: one block per (b, h, 128-row Q block)
  k_flash_attn<<<BB * HH * (SS / 128), blk, 0, stream>>>(qhd, khd, vtr, atn);

  // x2 = x + attn @ Wo^T
  k_gemm_bf16<<<dim3(DD / 128, BSTOK / 128), blk, 0, stream>>>(
      atn, wob, BSTOK, DD, DD, x2, nullptr, x, 1);

  // h2 = rmsnorm(x2) * ln2
  k_rmsnorm<<<BSTOK, blk, 0, stream>>>(x2, ln2, h2);

  // FFN
  k_gemm_bf16<<<dim3(DFFN / 128, BSTOK / 128), blk, 0, stream>>>(
      h2, w1b, BSTOK, DFFN, DD, nullptr, gg, nullptr, 2);
  k_gemm_bf16<<<dim3(DFFN / 128, BSTOK / 128), blk, 0, stream>>>(
      h2, w3b, BSTOK, DFFN, DD, nullptr, uu, nullptr, 2);
  k_silu_mul<<<16384, blk, 0, stream>>>(gg, uu, fi, (int)nTokF);

  // out = x2 + (silu*up) @ W2^T
  k_gemm_bf16<<<dim3(DD / 128, BSTOK / 128), blk, 0, stream>>>(
      fi, w2b, BSTOK, DD, DFFN, out, nullptr, x2, 1);
}
